// FocusedLinearAttention_21251498181313
// MI455X (gfx1250) — matrix operand order repaired and checked by reading, not compiled, not run on hardware
//
#include <hip/hip_runtime.h>
#include <hip/hip_bf16.h>
#include <math.h>

// ---------------------------------------------------------------------------
// Problem constants (from reference): B=8, N=4096, C=512, H=8, HD=64
// ---------------------------------------------------------------------------
#define BB   8
#define NN   4096
#define CC   512
#define HH   8
#define HD   64
#define BH   (BB*HH)           // 64
#define MROWS (BB*NN)          // 32768
#define NQKV  (3*CC)           // 1536 (q | k | v output columns)

typedef __attribute__((ext_vector_type(16))) _Float16 v16h;
typedef __attribute__((ext_vector_type(8)))  _Float16 v8h;
typedef __attribute__((ext_vector_type(4)))  _Float16 v4h;
typedef __attribute__((ext_vector_type(8)))  float    v8f;

// ---------------------------------------------------------------------------
// WMMA fragment loaders (wave32, ISA 7.12.2 layouts).
// A 16x32 f16 (MxK), row-major source: lane m=lane&15, g=lane>>4.
//   e<8  -> K = k0 + g*8 + e        (contiguous 8 halves -> b128)
//   e>=8 -> K = k0 + 16 + g*8 + e-8 (contiguous 8 halves -> b128)
// ---------------------------------------------------------------------------
__device__ __forceinline__ v16h load_fragA(const _Float16* __restrict__ A, int lda,
                                           int row0, int k0, int lane) {
  int m = lane & 15, g = lane >> 4;
  const _Float16* base = A + (size_t)(row0 + m) * lda + k0 + g * 8;
  v8h lo = *(const v8h*)(base);
  v8h hi = *(const v8h*)(base + 16);
  return __builtin_shufflevector(lo, hi, 0, 1, 2, 3, 4, 5, 6, 7,
                                 8, 9, 10, 11, 12, 13, 14, 15);
}

// B 32x16 f16 (KxN) from a TRANSPOSED source BT[N][K]:
// lane n=lane&15, g=lane>>4; e<8 -> K = k0 + g*8 + e, e>=8 -> K = k0 + 16 + g*8 + e-8
__device__ __forceinline__ v16h load_fragB_T(const _Float16* __restrict__ BT, int ldk,
                                             int k0, int col0, int lane) {
  int n = lane & 15, g = lane >> 4;
  const _Float16* base = BT + (size_t)(col0 + n) * ldk + k0 + g * 8;
  v8h lo = *(const v8h*)(base);
  v8h hi = *(const v8h*)(base + 16);
  return __builtin_shufflevector(lo, hi, 0, 1, 2, 3, 4, 5, 6, 7,
                                 8, 9, 10, 11, 12, 13, 14, 15);
}

#define WMMA_F16(acc, a, b) \
  __builtin_amdgcn_wmma_f32_16x16x32_f16(false, (a), false, (b), (short)0, (acc), false, false)

// ---------------------------------------------------------------------------
// K0: pack weights to f16 TRANSPOSED: wallT[NQKV][CC] = [Wq|Wkv]^T,
//     wprojT[CC][CC] = Wproj^T, invscale = 1/softplus(scale)
// ---------------------------------------------------------------------------
__global__ __launch_bounds__(256) void pack_kernel(
    const float* __restrict__ Wq, const float* __restrict__ Wkv,
    const float* __restrict__ Wproj, const float* __restrict__ scale_param,
    _Float16* __restrict__ wallT, _Float16* __restrict__ wprojT,
    float* __restrict__ invscale) {
  size_t idx = (size_t)blockIdx.x * 256 + threadIdx.x;
  const size_t n1 = (size_t)CC * NQKV;       // 512*1536
  const size_t n2 = (size_t)CC * CC;         // 512*512
  if (idx < n1) {
    int r = (int)(idx / NQKV), c = (int)(idx % NQKV);
    float v = (c < CC) ? Wq[(size_t)r * CC + c] : Wkv[(size_t)r * (2 * CC) + (c - CC)];
    wallT[(size_t)c * CC + r] = (_Float16)v;
  } else if (idx < n1 + n2) {
    size_t j = idx - n1;
    int r = (int)(j / CC), c = (int)(j % CC);
    wprojT[(size_t)c * CC + r] = (_Float16)Wproj[j];
  } else if (idx < n1 + n2 + CC) {
    int c = (int)(idx - n1 - n2);
    float sp = log1pf(expf(scale_param[c]));  // softplus
    invscale[c] = 1.0f / sp;
  }
}

// K1: convert x fp32 -> f16
__global__ __launch_bounds__(256) void cvt_kernel(const float* __restrict__ x,
                                                  _Float16* __restrict__ xh) {
  size_t i = (size_t)blockIdx.x * 256 + threadIdx.x;
  if (i < (size_t)MROWS * CC) xh[i] = (_Float16)x[i];
}

// ---------------------------------------------------------------------------
// K2: qkv = x @ [Wq|Wkv]  (32768 x 512 x 1536), WMMA, B transposed.
// Each wave computes a 16x64 strip (4 accumulators, A reused 4x).
// epilogue: q,k -> (relu+1e-6)*invscale (f16);  v -> f16
// ---------------------------------------------------------------------------
__global__ __launch_bounds__(256) void gemm_qkv_kernel(
    const _Float16* __restrict__ A, const _Float16* __restrict__ BT,
    const float* __restrict__ invscale,
    _Float16* __restrict__ qb, _Float16* __restrict__ kb, _Float16* __restrict__ vb) {
  const int lane = threadIdx.x & 31;
  const int wave = threadIdx.x >> 5;
  const int job  = blockIdx.x * 8 + wave;     // jobs = 2048 * 24
  const int ngrp = NQKV / 64;                 // 24
  const int mt = job / ngrp, ng = job % ngrp;
  const int row0 = mt * 16, col0 = ng * 64;
  v8f acc0 = {}, acc1 = {}, acc2 = {}, acc3 = {};
  for (int k0 = 0; k0 < CC; k0 += 32) {
    v16h a  = load_fragA(A, CC, row0, k0, lane);
    v16h b0 = load_fragB_T(BT, CC, k0, col0 +  0, lane);
    v16h b1 = load_fragB_T(BT, CC, k0, col0 + 16, lane);
    v16h b2 = load_fragB_T(BT, CC, k0, col0 + 32, lane);
    v16h b3 = load_fragB_T(BT, CC, k0, col0 + 48, lane);
    acc0 = WMMA_F16(acc0, a, b0);
    acc1 = WMMA_F16(acc1, a, b1);
    acc2 = WMMA_F16(acc2, a, b2);
    acc3 = WMMA_F16(acc3, a, b3);
  }
  const int g = lane >> 4, n = lane & 15;
  v8f accs[4] = {acc0, acc1, acc2, acc3};
#pragma unroll
  for (int nt = 0; nt < 4; ++nt) {
    const int col = col0 + nt * 16 + n;
#pragma unroll
    for (int r = 0; r < 8; ++r) {
      int row = row0 + g * 8 + r;
      float v = accs[nt][r];
      if (col < CC) {
        float t = (v > 0.f ? v : 0.f) + 1e-6f;
        qb[(size_t)row * CC + col] = (_Float16)(t * invscale[col]);
      } else if (col < 2 * CC) {
        float t = (v > 0.f ? v : 0.f) + 1e-6f;
        kb[(size_t)row * CC + (col - CC)] = (_Float16)(t * invscale[col - CC]);
      } else {
        vb[(size_t)row * CC + (col - 2 * CC)] = (_Float16)v;
      }
    }
  }
}

// ---------------------------------------------------------------------------
// K3: in-place focus per row (512 channels): t -> t^3 * ||t|| / ||t^3||
// ---------------------------------------------------------------------------
__global__ __launch_bounds__(256) void focus_kernel(_Float16* __restrict__ t) {
  const int row = blockIdx.x;
  const int tid = threadIdx.x;
  _Float16* p = t + (size_t)row * CC;
  float v0 = (float)p[tid], v1 = (float)p[tid + 256];
  float p0 = v0 * v0 * v0, p1 = v1 * v1 * v1;
  __shared__ float r2[256], r6[256];
  r2[tid] = v0 * v0 + v1 * v1;
  r6[tid] = p0 * p0 + p1 * p1;
  __syncthreads();
#pragma unroll
  for (int s = 128; s > 0; s >>= 1) {
    if (tid < s) { r2[tid] += r2[tid + s]; r6[tid] += r6[tid + s]; }
    __syncthreads();
  }
  float f = sqrtf(r2[0] / r6[0]);     // ||t|| / ||t^3||  (t>0 always)
  p[tid]       = (_Float16)(p0 * f);
  p[tid + 256] = (_Float16)(p1 * f);
}

// ---------------------------------------------------------------------------
// K4: ksum[bh][c] = sum_j k_f[bh, j, c]
// ---------------------------------------------------------------------------
__global__ __launch_bounds__(256) void ksum_kernel(const _Float16* __restrict__ kf,
                                                   float* __restrict__ ksum) {
  const int bh = blockIdx.x, tid = threadIdx.x;
  const int b = bh >> 3, h = bh & 7;
  const int c = tid & 63, q4 = tid >> 6;
  const _Float16* base = kf + ((size_t)b * NN) * CC + h * HD + c;
  float s = 0.f;
  for (int j = q4 * (NN / 4); j < (q4 + 1) * (NN / 4); ++j)
    s += (float)base[(size_t)j * CC];
  __shared__ float red[256];
  red[tid] = s;
  __syncthreads();
  if (q4 == 0)
    ksum[(size_t)bh * HD + c] = red[c] + red[64 + c] + red[128 + c] + red[192 + c];
}

// K5: rowz[bh*N+i] = 1 / (q_f[bh,i,:] . ksum[bh,:] + 1e-6)
__global__ __launch_bounds__(256) void rowz_kernel(const _Float16* __restrict__ qf,
                                                   const float* __restrict__ ksum,
                                                   float* __restrict__ rowz) {
  size_t idx = (size_t)blockIdx.x * 256 + threadIdx.x;
  if (idx >= (size_t)BH * NN) return;
  const int bh = (int)(idx >> 12), i = (int)(idx & (NN - 1));
  const int b = bh >> 3, h = bh & 7;
  const _Float16* q = qf + ((size_t)b * NN + i) * CC + h * HD;
  const float* ks = ksum + (size_t)bh * HD;
  float s = 0.f;
#pragma unroll
  for (int d = 0; d < HD; ++d) s += (float)q[d] * ks[d];
  rowz[idx] = 1.0f / (s + 1e-6f);
}

// ---------------------------------------------------------------------------
// K6: kvmT[bh] = (K^T V)^T per head, LDS-staged WMMA.
// K and V tiles staged TRANSPOSED ([ch][row]) so both fragments are
// contiguous vector LDS loads. Output stored transposed: kvmT[d][c].
// ---------------------------------------------------------------------------
__global__ __launch_bounds__(512) void kvm_kernel(const _Float16* __restrict__ kf,
                                                  const _Float16* __restrict__ vf,
                                                  _Float16* __restrict__ kvmT) {
  const int bh = blockIdx.x;
  const int b = bh >> 3, h = bh & 7;
  const int tid = threadIdx.x;
  const int wave = tid >> 5, lane = tid & 31;
  const int mt = wave >> 2, nt = wave & 3;
  __shared__ _Float16 ktT[64 * 32];   // [ch][row]
  __shared__ _Float16 vtT[64 * 32];   // [ch][row]
  const _Float16* kbase = kf + ((size_t)b * NN) * CC + h * HD;
  const _Float16* vbase = vf + ((size_t)b * NN) * CC + h * HD;
  const int r  = tid >> 4;          // 0..31 row of tile
  const int c4 = (tid & 15) * 4;    // 4 channels per thread
  v8f acc = {};
  for (int j0 = 0; j0 < NN; j0 += 32) {
    __syncthreads();
    v4h kv4 = *(const v4h*)&kbase[(size_t)(j0 + r) * CC + c4];
    v4h vv4 = *(const v4h*)&vbase[(size_t)(j0 + r) * CC + c4];
#pragma unroll
    for (int j = 0; j < 4; ++j) {
      ktT[(c4 + j) * 32 + r] = kv4[j];
      vtT[(c4 + j) * 32 + r] = vv4[j];
    }
    __syncthreads();
    // A = K^T: M=channel, K=row -> ktT is "row-major" for A with lda=32
    v16h a  = load_fragA(ktT, 32, mt * 16, 0, lane);
    v16h bf = load_fragB_T(vtT, 32, 0, nt * 16, lane);
    acc = WMMA_F16(acc, a, bf);
  }
  const int g = lane >> 4, n = lane & 15;
#pragma unroll
  for (int rr = 0; rr < 8; ++rr) {
    int c = mt * 16 + g * 8 + rr;   // row of kvm (channel)
    int d = nt * 16 + n;            // col of kvm
    kvmT[((size_t)bh * HD + d) * HD + c] = (_Float16)acc[rr];
  }
}

// ---------------------------------------------------------------------------
// K7: depthwise 5x5 conv (SAME) on q_f viewed as (bh, 64, 64, HD) + bias
// ---------------------------------------------------------------------------
__global__ __launch_bounds__(256) void dwconv_kernel(const _Float16* __restrict__ qf,
                                                     const float* __restrict__ w,
                                                     const float* __restrict__ bias,
                                                     _Float16* __restrict__ convo) {
  size_t idx = (size_t)blockIdx.x * 256 + threadIdx.x;
  if (idx >= (size_t)BH * NN * HD) return;
  const int c  = (int)(idx & 63);
  const int n  = (int)((idx >> 6) & (NN - 1));
  const int bh = (int)(idx >> 18);
  const int y = n >> 6, x = n & 63;
  const int b = bh >> 3, h = bh & 7;
  const _Float16* src = qf + ((size_t)b * NN) * CC + h * HD + c;
  float acc = bias[c];
#pragma unroll
  for (int ky = 0; ky < 5; ++ky) {
    int yy = y + ky - 2;
    if (yy < 0 || yy > 63) continue;
#pragma unroll
    for (int kx = 0; kx < 5; ++kx) {
      int xx = x + kx - 2;
      if (xx < 0 || xx > 63) continue;
      acc += w[c * 25 + ky * 5 + kx] * (float)src[(size_t)(yy * 64 + xx) * CC];
    }
  }
  convo[idx] = (_Float16)acc;
}

// ---------------------------------------------------------------------------
// K8: out = (q_f @ kvm) * z + conv, per head, WMMA; each wave does 16x64
// grid = 64 bh * 32 blocks; block = 256 (8 waves)
// ---------------------------------------------------------------------------
__global__ __launch_bounds__(256) void out_gemm_kernel(
    const _Float16* __restrict__ qf, const _Float16* __restrict__ kvmT,
    const float* __restrict__ rowz, const _Float16* __restrict__ convo,
    _Float16* __restrict__ outp) {
  const int bh = blockIdx.x >> 5;
  const int blk = blockIdx.x & 31;
  const int wave = threadIdx.x >> 5, lane = threadIdx.x & 31;
  const int mt = blk * 8 + wave;          // 0..255
  const int b = bh >> 3, h = bh & 7;
  const _Float16* Abase = qf + ((size_t)b * NN) * CC + h * HD;
  const _Float16* BTbase = kvmT + (size_t)bh * HD * HD;
  v8f acc0 = {}, acc1 = {}, acc2 = {}, acc3 = {};
#pragma unroll
  for (int k0 = 0; k0 < HD; k0 += 32) {
    v16h a  = load_fragA(Abase, CC, mt * 16, k0, lane);
    v16h b0 = load_fragB_T(BTbase, HD, k0,  0, lane);
    v16h b1 = load_fragB_T(BTbase, HD, k0, 16, lane);
    v16h b2 = load_fragB_T(BTbase, HD, k0, 32, lane);
    v16h b3 = load_fragB_T(BTbase, HD, k0, 48, lane);
    acc0 = WMMA_F16(acc0, a, b0);
    acc1 = WMMA_F16(acc1, a, b1);
    acc2 = WMMA_F16(acc2, a, b2);
    acc3 = WMMA_F16(acc3, a, b3);
  }
  const int g = lane >> 4, n = lane & 15;
  v8f accs[4] = {acc0, acc1, acc2, acc3};
#pragma unroll
  for (int nt = 0; nt < 4; ++nt) {
    const int d = nt * 16 + n;
#pragma unroll
    for (int r = 0; r < 8; ++r) {
      int i = mt * 16 + g * 8 + r;
      float val = accs[nt][r] * rowz[(size_t)bh * NN + i] +
                  (float)convo[((size_t)bh * NN + i) * HD + d];
      outp[((size_t)b * NN + i) * CC + h * HD + d] = (_Float16)val;
    }
  }
}

// ---------------------------------------------------------------------------
// K9: y = outp @ Wproj + b_proj  (32768 x 512 x 512), fp32 out, B transposed
// ---------------------------------------------------------------------------
__global__ __launch_bounds__(256) void gemm_proj_kernel(
    const _Float16* __restrict__ A, const _Float16* __restrict__ BT,
    const float* __restrict__ bias, float* __restrict__ out) {
  const int lane = threadIdx.x & 31;
  const int wave = threadIdx.x >> 5;
  const int job = blockIdx.x * 8 + wave;    // 2048 * 8 jobs
  const int ngrp = CC / 64;                 // 8
  const int mt = job / ngrp, ng = job % ngrp;
  const int row0 = mt * 16, col0 = ng * 64;
  v8f acc0 = {}, acc1 = {}, acc2 = {}, acc3 = {};
  for (int k0 = 0; k0 < CC; k0 += 32) {
    v16h a  = load_fragA(A, CC, row0, k0, lane);
    v16h b0 = load_fragB_T(BT, CC, k0, col0 +  0, lane);
    v16h b1 = load_fragB_T(BT, CC, k0, col0 + 16, lane);
    v16h b2 = load_fragB_T(BT, CC, k0, col0 + 32, lane);
    v16h b3 = load_fragB_T(BT, CC, k0, col0 + 48, lane);
    acc0 = WMMA_F16(acc0, a, b0);
    acc1 = WMMA_F16(acc1, a, b1);
    acc2 = WMMA_F16(acc2, a, b2);
    acc3 = WMMA_F16(acc3, a, b3);
  }
  const int g = lane >> 4, n = lane & 15;
  v8f accs[4] = {acc0, acc1, acc2, acc3};
#pragma unroll
  for (int nt = 0; nt < 4; ++nt) {
    const int col = col0 + nt * 16 + n;
    const float bp = bias[col];
#pragma unroll
    for (int r = 0; r < 8; ++r)
      out[(size_t)(row0 + g * 8 + r) * CC + col] = accs[nt][r] + bp;
  }
}

// ---------------------------------------------------------------------------
extern "C" void kernel_launch(void* const* d_in, const int* in_sizes, int n_in,
                              void* d_out, int out_size, void* d_ws, size_t ws_size,
                              hipStream_t stream) {
  const float* x          = (const float*)d_in[0];
  const float* Wq         = (const float*)d_in[1];
  const float* Wkv        = (const float*)d_in[2];
  const float* Wproj      = (const float*)d_in[3];
  const float* b_proj     = (const float*)d_in[4];
  const float* scale_p    = (const float*)d_in[5];
  const float* dwc_w      = (const float*)d_in[6];
  const float* dwc_b      = (const float*)d_in[7];
  float* out = (float*)d_out;

  // workspace carve-up (256B aligned)
  size_t off = 0;
  auto carve = [&](size_t bytes) {
    void* p = (char*)d_ws + off;
    off = (off + bytes + 255) & ~(size_t)255;
    return p;
  };
  _Float16* xh     = (_Float16*)carve((size_t)MROWS * CC * 2);     // 33.5MB
  _Float16* wallT  = (_Float16*)carve((size_t)CC * NQKV * 2);      // 1.5MB
  _Float16* wprojT = (_Float16*)carve((size_t)CC * CC * 2);        // 0.5MB
  float*    invsc  = (float*)   carve((size_t)CC * 4);
  _Float16* qb     = (_Float16*)carve((size_t)MROWS * CC * 2);     // 33.5MB
  _Float16* kb     = (_Float16*)carve((size_t)MROWS * CC * 2);     // 33.5MB
  _Float16* vb     = (_Float16*)carve((size_t)MROWS * CC * 2);     // 33.5MB
  float*    ksum   = (float*)   carve((size_t)BH * HD * 4);
  float*    rowz   = (float*)   carve((size_t)BH * NN * 4);        // 1MB
  _Float16* kvmT   = (_Float16*)carve((size_t)BH * HD * HD * 2);   // 0.5MB
  _Float16* convo  = (_Float16*)carve((size_t)BH * NN * HD * 2);   // 33.5MB
  _Float16* outp   = (_Float16*)carve((size_t)MROWS * CC * 2);     // 33.5MB
  (void)ws_size; (void)in_sizes; (void)n_in; (void)out_size;

  // 0: pack (transposed) weights + invscale
  {
    size_t tot = (size_t)CC * NQKV + (size_t)CC * CC + CC;
    pack_kernel<<<(unsigned)((tot + 255) / 256), 256, 0, stream>>>(
        Wq, Wkv, Wproj, scale_p, wallT, wprojT, invsc);
  }
  // 1: x -> f16
  cvt_kernel<<<(unsigned)(((size_t)MROWS * CC + 255) / 256), 256, 0, stream>>>(x, xh);
  // 2: qkv GEMM (WMMA)
  gemm_qkv_kernel<<<(2048 * (NQKV / 64)) / 8, 256, 0, stream>>>(xh, wallT, invsc, qb, kb, vb);
  // 3: focus q and k (in place)
  focus_kernel<<<MROWS, 256, 0, stream>>>(qb);
  focus_kernel<<<MROWS, 256, 0, stream>>>(kb);
  // 4: ksum
  ksum_kernel<<<BH, 256, 0, stream>>>(kb, ksum);
  // 5: rowz
  rowz_kernel<<<(BH * NN) / 256, 256, 0, stream>>>(qb, ksum, rowz);
  // 6: kvm = (K^T V)^T (WMMA)
  kvm_kernel<<<BH, 512, 0, stream>>>(kb, vb, kvmT);
  // 7: depthwise conv
  dwconv_kernel<<<(unsigned)(((size_t)BH * NN * HD + 255) / 256), 256, 0, stream>>>(
      qb, dwc_w, dwc_b, convo);
  // 8: out = q@kvm * z + conv (WMMA)
  out_gemm_kernel<<<BH * 32, 256, 0, stream>>>(qb, kvmT, rowz, convo, outp);
  // 9: projection (WMMA) -> fp32 out
  gemm_proj_kernel<<<(2048 * (CC / 64)) / 8, 256, 0, stream>>>(outp, wprojT, b_proj, out);
}
